// LSTM_71416716198165
// MI455X (gfx1250) — matrix operand order repaired and checked by reading, not compiled, not run on hardware
//
#include <hip/hip_runtime.h>
#include <hip/hip_bf16.h>

#define SEQ     512
#define HID     256
#define NBATCH  256
#define BT      16          // batch columns per workgroup
#define NSTEP   (SEQ - 1)   // 511 recurrence steps

typedef __attribute__((ext_vector_type(16))) __bf16        v16bf;
typedef __attribute__((ext_vector_type(8)))  float         v8f;
typedef __attribute__((ext_vector_type(8)))  unsigned int  v8u;

// ---------- helpers ----------
__device__ __forceinline__ unsigned int f2bf_bits(float f) {
    unsigned int u = __float_as_uint(f);
    unsigned int r = u + 0x7FFFu + ((u >> 16) & 1u);
    return r >> 16;
}
__device__ __forceinline__ float bf2f(unsigned int b) {
    return __uint_as_float(b << 16);
}
__device__ __forceinline__ float fast_rcp(float x) { return __builtin_amdgcn_rcpf(x); }
__device__ __forceinline__ float fast_sig(float x) { return fast_rcp(1.0f + __expf(-x)); }
__device__ __forceinline__ float fast_tanh(float x) { return 2.0f * fast_rcp(1.0f + __expf(-2.0f * x)) - 1.0f; }

// ------------------------------------------------------------------
// Kernel 0: pack gh/ih/fh/oh (fp32 row-major HxH) into bf16 WMMA
// A-fragment layout: byte = ((g*16+MT)*8+kt)*1024 + lane*32 + e*2
// A 16x32 16-bit frag, this file's (contiguous) k order: M = lane%16; hi = lane/16; e = v*2+half;
//   K = hi*16 + e
// ------------------------------------------------------------------
__global__ __launch_bounds__(256) void pack_weights(
        const float* __restrict__ gh, const float* __restrict__ ih,
        const float* __restrict__ fh, const float* __restrict__ oh,
        unsigned short* __restrict__ wpack) {
    int tid = blockIdx.x * 256 + threadIdx.x;      // 0 .. 262143
    int e   = tid & 15;
    int l   = (tid >> 4) & 31;
    int kt  = (tid >> 9) & 7;
    int MT  = (tid >> 12) & 15;
    int g   = (tid >> 16) & 3;

    const float* W = (g == 0) ? gh : (g == 1) ? ih : (g == 2) ? fh : oh;

    int m    = MT * 16 + (l & 15);
    int hi   = l >> 4;
    int v    = e >> 1;
    int half = e & 1;
    int kk   = hi * 16 + v * 2 + half;
    int k    = kt * 32 + kk;

    wpack[tid] = (unsigned short)f2bf_bits(W[m * HID + k]);
}

// ------------------------------------------------------------------
// Kernel 1: 511-step LSTM recurrence.
// 16 WGs (one per 16 batch cols) x 256 threads (8 waves).
// Wave w owns hidden rows [w*32, w*32+32) for all 4 gates.
// Weight A-fragments stream L2 -> LDS via GLOBAL_LOAD_ASYNC_TO_LDS_B128
// into a per-wave 4-slot staging ring at issue distance 2 (zero VGPR
// in-flight cost, ASYNCcnt-tracked, in-order completion => uniform
// s_wait_asynccnt 32), then ds_load_b128 -> v_wmma_f32_16x16x32_bf16.
// h double-buffered in LDS in B-fragment layout (one barrier per step).
// ------------------------------------------------------------------
__global__ __launch_bounds__(256, 1) void lstm_recurrence(
        const int*   __restrict__ x,     const float* __restrict__ emb,
        const float* __restrict__ gxv,   const float* __restrict__ gbv,
        const float* __restrict__ ixv,   const float* __restrict__ ibv,
        const float* __restrict__ fxv,   const float* __restrict__ fbv,
        const float* __restrict__ oxv,   const float* __restrict__ obv,
        const unsigned short* __restrict__ wpack,
        float* __restrict__ hfinal) {

    // per-wave staging ring: wstage[wave][slot(4)][j(8)][lane(32)][32B] = 256 KB
    __shared__ alignas(16) unsigned char  wstage[8 * 4 * 8192];
    __shared__ alignas(16) unsigned short hbuf[2][8 * 512];       // 16 KB h B-frags

    const int tid  = threadIdx.x;
    const int wave = tid >> 5;        // 0..7
    const int lane = tid & 31;
    const int n    = lane & 15;       // batch column within tile
    const int hi   = lane >> 4;       // lane half
    const int b0   = blockIdx.x * BT;

    // ---- invariant async-copy address registers ----
    const unsigned lane32 = (unsigned)lane * 32u;
    // LDS dest base for this wave's slot 0 (per-lane byte address, low 32 of flat)
    const unsigned vdst0 =
        (unsigned)(size_t)(&wstage[wave * 32768]) + lane32;
    // global offset VGPRs per fragment j=(g,mt): lane*32 + g*129024 + mt*7168
    unsigned vaj[8];
#pragma unroll
    for (int j = 0; j < 8; ++j) {
        const int g = j >> 1, mt = j & 1;
        vaj[j] = lane32 + (unsigned)(g * 129024 + mt * 7168);
    }
    // uniform global base for this wave (kt added per slice)
    const unsigned char* wbase = (const unsigned char*)wpack + wave * 16384;

    // ---- zero the t=0 read buffer (h0 = 0) ----
    for (int idx = tid; idx < 8 * 512; idx += 256) hbuf[0][idx] = 0;
    __syncthreads();

    // ---- per-lane step-invariant coefficients, packed bf16: lo=cx, hi=cb ----
    const float* xs_[4] = {gxv, ixv, fxv, oxv};
    const float* bs_[4] = {gbv, ibv, fbv, obv};
    unsigned int cxb[4][2][8];
#pragma unroll
    for (int g = 0; g < 4; ++g)
#pragma unroll
        for (int mt = 0; mt < 2; ++mt)
#pragma unroll
            for (int r = 0; r < 8; ++r) {
                int m = wave * 32 + mt * 16 + r + 8 * hi;   // D-layout row
                cxb[g][mt][r] = f2bf_bits(xs_[g][m]) | (f2bf_bits(bs_[g][m]) << 16);
            }

    // ---- cell state entirely in registers ----
    float cst[2][8];
#pragma unroll
    for (int mt = 0; mt < 2; ++mt)
#pragma unroll
        for (int r = 0; r < 8; ++r) cst[mt][r] = 0.0f;

    // async-issue one kt-slice (16 x B128; IOFFSET = j*1024 + half*16 adds to
    // BOTH lds and global sides; asymmetric strides live in SADDR/VADDR/VDST)
    auto issue_slice = [&](int kt, int slot) {
        const unsigned char* sa = wbase + (unsigned)kt * 1024u;   // SGPR pair
        const unsigned vd = vdst0 + (unsigned)slot * 8192u;
#pragma unroll
        for (int j = 0; j < 8; ++j) {
            asm volatile("global_load_async_to_lds_b128 %0, %1, %2 offset:%3"
                         :: "v"(vd), "v"(vaj[j]), "s"(sa), "n"(j * 1024)
                         : "memory");
            asm volatile("global_load_async_to_lds_b128 %0, %1, %2 offset:%3"
                         :: "v"(vd), "v"(vaj[j]), "s"(sa), "n"(j * 1024 + 16)
                         : "memory");
        }
    };

    const v8f zero8 = {0.f, 0.f, 0.f, 0.f, 0.f, 0.f, 0.f, 0.f};

    // prologue: prefetch slices 0,1 into slots 0,1 (issue distance 2)
    issue_slice(0, 0);
    issue_slice(1, 1);

    // ================= recurrence =================
#pragma unroll 1
    for (int t = 0; t < NSTEP; ++t) {
        const int rb = t & 1;         // read buffer
        const int wb = rb ^ 1;        // write buffer

        // per-step input term (chained global loads; hidden behind the slices)
        const float xt = emb[x[(b0 + n) * SEQ + t]];

        v8f acc[4][2];
#pragma unroll
        for (int g = 0; g < 4; ++g)
#pragma unroll
            for (int mt = 0; mt < 2; ++mt) acc[g][mt] = zero8;

#pragma unroll
        for (int kt = 0; kt < 8; ++kt) {
            // all prior ds reads landed -> safe to overwrite the target slot
            asm volatile("s_wait_dscnt 0" ::: "memory");
            // prefetch slice kt+2 (wraps into next step across the boundary)
            issue_slice((kt + 2) & 7, (kt + 2) & 3);
            // asyncs complete in order: <=32 outstanding => slice kt has landed
            asm volatile("s_wait_asynccnt %0" :: "n"(32) : "memory");

            // consume slice kt from slot kt&3
            v8u   braw  = *(const v8u*)(&hbuf[rb][kt * 512 + lane * 16]);
            v16bf bfrag = __builtin_bit_cast(v16bf, braw);
            const unsigned char* sl =
                &wstage[wave * 32768 + (kt & 3) * 8192 + lane * 32];
#pragma unroll
            for (int j = 0; j < 8; ++j) {
                const int g = j >> 1, mt = j & 1;
                v16bf af = __builtin_bit_cast(v16bf, *(const v8u*)(sl + j * 1024));
                acc[g][mt] = __builtin_amdgcn_wmma_f32_16x16x32_bf16(
                    false, af, false, bfrag, (short)0, acc[g][mt], false, false);
            }
        }

        // gate combine (wave-local) + publish h_new as bf16 B-fragments
#pragma unroll
        for (int mt = 0; mt < 2; ++mt) {
            unsigned int pk[4];
#pragma unroll
            for (int r = 0; r < 8; ++r) {
                float cxg = __uint_as_float(cxb[0][mt][r] << 16);
                float cbg = __uint_as_float(cxb[0][mt][r] & 0xffff0000u);
                float cxi = __uint_as_float(cxb[1][mt][r] << 16);
                float cbi = __uint_as_float(cxb[1][mt][r] & 0xffff0000u);
                float cxf = __uint_as_float(cxb[2][mt][r] << 16);
                float cbf = __uint_as_float(cxb[2][mt][r] & 0xffff0000u);
                float cxo = __uint_as_float(cxb[3][mt][r] << 16);
                float cbo = __uint_as_float(cxb[3][mt][r] & 0xffff0000u);

                float pg = acc[0][mt][r] + cxg * xt + cbg;
                float pi = acc[1][mt][r] + cxi * xt + cbi;
                float pf = acc[2][mt][r] + cxf * xt + cbf;
                float po = acc[3][mt][r] + cxo * xt + cbo;
                float gg = fast_tanh(pg);
                float ii = fast_sig(pi);
                float ff = fast_sig(pf);
                float oo = fast_sig(po);
                float c  = gg * ii + cst[mt][r] * ff;
                cst[mt][r] = c;
                float h  = fast_tanh(c) * oo;

                unsigned int hb = f2bf_bits(h);
                if ((r & 1) == 0) pk[r >> 1] = hb;
                else              pk[r >> 1] |= hb << 16;
            }
            // 8 bf16 values contiguous in B-fragment layout (16B aligned)
            int off = wave * 512 + (mt * 16 + n) * 16 + 8 * hi;
            uint4 v4; v4.x = pk[0]; v4.y = pk[1]; v4.z = pk[2]; v4.w = pk[3];
            *(uint4*)(&hbuf[wb][off]) = v4;
        }
        __syncthreads();   // writes to wb complete before next step reads it
    }

    // ---- final h lives in hbuf[1] (t=510 wrote wb=1): coalesced export ----
    for (int idx = tid; idx < 8 * 512; idx += 256) {
        int kt    = idx >> 9;
        int rest  = idx & 511;
        int laneb = rest >> 4;
        int e     = rest & 15;
        int row   = kt * 32 + (laneb >> 4) * 16 + e;
        int col   = laneb & 15;
        hfinal[row * NBATCH + b0 + col] = bf2f(hbuf[1][idx]);
    }
}

// ------------------------------------------------------------------
// Kernel 2: p = ph @ h + pb ; softmax over batch axis; out[b*256+k]
// ------------------------------------------------------------------
__global__ __launch_bounds__(256) void proj_softmax(
        const float* __restrict__ ph, const float* __restrict__ pb,
        const float* __restrict__ hfinal, float* __restrict__ out) {
    const int k = blockIdx.x;
    const int b = threadIdx.x;

    float acc = pb[k];
    for (int j = 0; j < HID; ++j)
        acc += ph[k * HID + j] * hfinal[j * NBATCH + b];   // ph[k,j] uniform -> SMEM

    __shared__ float red[256];
    red[b] = acc;
    __syncthreads();
    for (int s = 128; s > 0; s >>= 1) {
        if (b < s) red[b] = fmaxf(red[b], red[b + s]);
        __syncthreads();
    }
    float mx = red[0];
    __syncthreads();

    float e = __expf(acc - mx);
    red[b] = e;
    __syncthreads();
    for (int s = 128; s > 0; s >>= 1) {
        if (b < s) red[b] = red[b] + red[b + s];
        __syncthreads();
    }
    float inv = 1.0f / red[0];

    out[b * HID + k] = e * inv;
}

// ------------------------------------------------------------------
extern "C" void kernel_launch(void* const* d_in, const int* in_sizes, int n_in,
                              void* d_out, int out_size, void* d_ws, size_t ws_size,
                              hipStream_t stream) {
    const int*   x   = (const int*)  d_in[0];
    const float* emb = (const float*)d_in[1];
    const float* gx  = (const float*)d_in[2];
    const float* gh  = (const float*)d_in[3];
    const float* gb  = (const float*)d_in[4];
    const float* ix  = (const float*)d_in[5];
    const float* ih  = (const float*)d_in[6];
    const float* ib  = (const float*)d_in[7];
    const float* fx  = (const float*)d_in[8];
    const float* fh  = (const float*)d_in[9];
    const float* fb  = (const float*)d_in[10];
    const float* ox  = (const float*)d_in[11];
    const float* oh  = (const float*)d_in[12];
    const float* ob  = (const float*)d_in[13];
    const float* ph  = (const float*)d_in[14];
    const float* pb  = (const float*)d_in[15];

    unsigned short* wpack  = (unsigned short*)d_ws;                       // 512 KB
    float*          hfinal = (float*)((char*)d_ws +
                              (size_t)(4 * 16 * 8 * 512) * sizeof(unsigned short)); // 256 KB
    float*          out    = (float*)d_out;

    pack_weights<<<1024, 256, 0, stream>>>(gh, ih, fh, oh, wpack);
    lstm_recurrence<<<NBATCH / BT, 256, 0, stream>>>(
        x, emb, gx, gb, ix, ib, fx, fb, ox, ob, wpack, hfinal);
    proj_softmax<<<HID, 256, 0, stream>>>(ph, pb, hfinal, out);
}
